// PFC_7464653160448
// MI455X (gfx1250) — matrix operand order repaired and checked by reading, not compiled, not run on hardware
//
#include <hip/hip_runtime.h>

// ---------------------------------------------------------------------------
// CDNA5 / gfx1250: wave32, WMMA 16x16x32 bf16 GEMMs with pre-packed im2col.
// ---------------------------------------------------------------------------

typedef __attribute__((ext_vector_type(16))) __bf16 v16bf;
typedef __attribute__((ext_vector_type(8)))  __bf16 v8bf;
typedef __attribute__((ext_vector_type(8)))  float  v8f;

#define B_     2
#define N_     4
#define D_     256
#define P_     30000
#define GX_    60
#define GY_    60
#define GZ_    8
#define NIMG   8              // B_*N_
#define NROWS  60000          // B_*P_
#define NVOX   57600          // B_*GX_*GY_*GZ_
#define NPART  235            // ceil(NROWS/256)
#define K1_    12544          // 256*7*7
#define K2_    3136           // 64*7*7
#define K3_    576            // 64*3*3
#define FIXS      16384.0f
#define INV_FIXS  (1.0f/16384.0f)

// ---- generic one-wave bf16 GEMM tile: out = relu(A[16xK] * Bt[16xK]^T) -----
// A: [M][K] bf16 row-major.  Bt: per-image [N][K] bf16 (B transposed).
// grid: (M/16 * N/16, NIMG), block: 32 (one wave, EXEC all ones).
__global__ void gemm_relu_kernel(const __bf16* __restrict__ A,
                                 const __bf16* __restrict__ Bt,
                                 float* __restrict__ out,
                                 int K, int ntile, int N, int M) {
  const int img = blockIdx.y;
  const int mt  = (blockIdx.x / ntile) * 16;
  const int nt  = (blockIdx.x % ntile) * 16;
  const int lane = threadIdx.x & 31;
  const int hs   = lane >> 4;          // 0: lanes 0-15, 1: lanes 16-31
  const int rc   = lane & 15;          // A row / B col within tile

  const v8bf* __restrict__ Arow =
      (const v8bf*)(A + (size_t)(mt + rc) * K);
  const v8bf* __restrict__ Brow =
      (const v8bf*)(Bt + ((size_t)img * N + (nt + rc)) * K);

  v8f acc = {};
  const int K8 = K >> 3;
  for (int k8 = 0; k8 < K8; k8 += 4) {   // 32 K per iteration
    // A 16x32 fragment: lane holds K {hs*8+0..7, 16+hs*8+0..7}
    const v8bf a_lo = Arow[k8 + hs];
    const v8bf a_hi = Arow[k8 + 2 + hs];
    // B 32x16 fragment: lane holds K {hs*8+0..7, 16+hs*8+0..7}
    const v8bf b_lo = Brow[k8 + hs];
    const v8bf b_hi = Brow[k8 + 2 + hs];
    const v16bf a = __builtin_shufflevector(a_lo, a_hi,
        0,1,2,3,4,5,6,7,8,9,10,11,12,13,14,15);
    const v16bf b = __builtin_shufflevector(b_lo, b_hi,
        0,1,2,3,4,5,6,7,8,9,10,11,12,13,14,15);
    acc = __builtin_amdgcn_wmma_f32_16x16x32_bf16(false, a, false, b,
                                                  (short)0, acc, false, false);
  }

  float* __restrict__ Oimg = out + (size_t)img * M * N;
#pragma unroll
  for (int v = 0; v < 8; ++v) {          // C/D: VGPR v -> M = v (+8 hi lanes)
    const int m = mt + v + hs * 8;
    float val = acc[v];
    Oimg[(size_t)m * N + nt + rc] = val > 0.f ? val : 0.f;
  }
}

// ---- weight convert f32 -> bf16 (K-contiguous already) ---------------------
__global__ void cvt_bf16_kernel(const float* __restrict__ src,
                                __bf16* __restrict__ dst, int n) {
  const int i = blockIdx.x * 256 + threadIdx.x;
  if (i < n) dst[i] = (__bf16)src[i];
}

// ---- im2col pack for conv1 (7x7 pad3 on 32x32 clip; out rows 0..1) ---------
// grid (64 n, NIMG), block 256 over k.
__global__ void pack_b1_kernel(const float* __restrict__ clip,
                               __bf16* __restrict__ Bt) {
  const int img = blockIdx.y;
  const int n   = blockIdx.x;            // oy*32+ox, oy in {0,1}
  const int oy = n >> 5, ox = n & 31;
  const float* __restrict__ xin = clip + (size_t)img * D_ * 32 * 32;
  __bf16* __restrict__ dst = Bt + ((size_t)img * 64 + n) * K1_;
  for (int k = threadIdx.x; k < K1_; k += 256) {
    const int ci = k / 49, rr = k - ci * 49;
    const int fy = rr / 7, fx = rr - fy * 7;
    const int iy = oy - 3 + fy, ix = ox - 3 + fx;
    const float v = ((unsigned)iy < 32u && (unsigned)ix < 32u)
                        ? xin[(ci * 32 + iy) * 32 + ix] : 0.f;
    dst[k] = (__bf16)v;
  }
}

// ---- im2col pack for conv2 (7x7 pad3 on nearest-up2(x1); out row 0) --------
// grid (64 n, NIMG), block 256 over k.
__global__ void pack_b2_kernel(const float* __restrict__ x1,
                               __bf16* __restrict__ Bt) {
  const int img = blockIdx.y;
  const int n   = blockIdx.x;            // ox in 0..63
  const float* __restrict__ xin = x1 + (size_t)img * 64 * 64;
  __bf16* __restrict__ dst = Bt + ((size_t)img * 64 + n) * K2_;
  for (int k = threadIdx.x; k < K2_; k += 256) {
    const int ci = k / 49, rr = k - ci * 49;
    const int fy = rr / 7, fx = rr - fy * 7;
    const int iy = fy - 3;               // output row 0
    const int ix = n - 3 + fx;
    const float v = ((unsigned)iy < 64u && (unsigned)ix < 64u)
                        ? xin[ci * 64 + (iy >> 1) * 32 + (ix >> 1)] : 0.f;
    dst[k] = (__bf16)v;
  }
}

// ---- im2col pack for conv3 (3x3 pad1 on nearest-up2(x2); out row 0) --------
// grid (128 n, NIMG), block 256 over k (576 < 256*3).
__global__ void pack_b3_kernel(const float* __restrict__ x2,
                               __bf16* __restrict__ Bt) {
  const int img = blockIdx.y;
  const int n   = blockIdx.x;            // ox in 0..127
  const float* __restrict__ xin = x2 + (size_t)img * 64 * 64;
  __bf16* __restrict__ dst = Bt + ((size_t)img * 128 + n) * K3_;
  for (int k = threadIdx.x; k < K3_; k += 256) {
    const int ci = k / 9, rr = k - ci * 9;
    const int fy = rr / 3, fx = rr - fy * 3;
    const int iy = fy - 1;               // rows -1,0,1 -> x2 row 0
    const int ix = n - 1 + fx;
    const float v = ((unsigned)iy < 128u && (unsigned)ix < 128u)
                        ? xin[ci * 64 + (ix >> 1)] : 0.f;
    dst[k] = (__bf16)v;
  }
}

// ---- fmap row 0: bilinear align-corners x-upsample 128 -> 256 --------------
__global__ void upsample_row_kernel(const float* __restrict__ x3, float* __restrict__ F) {
  const int row = blockIdx.x;            // img*256 + d
  const int x   = threadIdx.x;
  const float pos = (float)x * (127.0f / 255.0f);
  const float j0f = floorf(pos);
  const int   j0  = (int)j0f;
  const int   j1  = (j0 + 1 < 128) ? j0 + 1 : 127;
  const float w   = pos - j0f;
  const float* src = x3 + (size_t)row * 128;
  F[(size_t)row * 256 + x] = (1.0f - w) * src[j0] + w * src[j1];
}

// ---- grid-sample + mean over views, with the reference's flat-reshape ------
__global__ void feats_kernel(const float* __restrict__ pc, const float* __restrict__ F,
                             float* __restrict__ feats) {
  const int bp   = blockIdx.x;
  const int b    = bp / P_;
  const int prow = bp - b * P_;
  const int dcol = threadIdx.x;
  const int q = prow * D_ + dcol;        // flat reinterpret of [D,P] block
  const int d = q / P_;
  const int p = q - d * P_;
  float acc = 0.f;
#pragma unroll
  for (int n = 0; n < N_; ++n) {
    const int img  = b * N_ + n;
    const float px = pc[((size_t)img * P_ + p) * 2];        // only x coord used
    const float gx = px * (1.0f / 256.0f) - 1.0f;
    const float gy = gx * (1.0f / 256.0f) - 1.0f;           // reference's y bug
    const float ixf = ((gx + 1.0f) * 256.0f - 1.0f) * 0.5f;
    const float iyf = ((gy + 1.0f) * 256.0f - 1.0f) * 0.5f; // in [-1,0)
    const float x0f = floorf(ixf);
    const float y0f = floorf(iyf);
    const float wx  = ixf - x0f;
    const float wy  = iyf - y0f;
    const int x0 = (int)x0f;
    const int y1 = (int)y0f + 1;
    float row = 0.f;
    if (y1 == 0) {                       // y0 corner invalid; y1 is row 0
      const float* Fr = F + ((size_t)img * D_ + d) * 256;
      const float v0 = (x0 >= 0 && x0 < 256) ? Fr[x0] : 0.f;
      const int   xc = x0 + 1;
      const float v1 = (xc >= 0 && xc < 256) ? Fr[xc] : 0.f;
      row = wy * ((1.0f - wx) * v0 + wx * v1);
    }
    acc += row;
  }
  feats[(size_t)bp * D_ + dcol] = acc * 0.25f;
}

// ---- deterministic BN stats: two-stage tree reduction ----------------------
__global__ void stats_partial(const float* __restrict__ feats,
                              float* __restrict__ pS, float* __restrict__ pQ) {
  const int blk = blockIdx.x;
  const int d   = threadIdx.x;
  float s = 0.f, qq = 0.f;
  const int r0 = blk * 256;
  for (int j = 0; j < 256; ++j) {
    const int r = r0 + j;
    if (r < NROWS) {
      const float f = feats[(size_t)r * D_ + d];
      s += f; qq += f * f;
    }
  }
  pS[(size_t)blk * D_ + d] = s;
  pQ[(size_t)blk * D_ + d] = qq;
}

__global__ void stats_final(const float* __restrict__ pS, const float* __restrict__ pQ,
                            const float* __restrict__ gamma,
                            float* __restrict__ mu, float* __restrict__ scl) {
  const int d = threadIdx.x;
  float s = 0.f, qq = 0.f;
  for (int i = 0; i < NPART; ++i) { s += pS[i * D_ + d]; qq += pQ[i * D_ + d]; }
  const float m = s * (1.0f / NROWS);
  float var = qq * (1.0f / NROWS) - m * m;
  var = var > 0.f ? var : 0.f;
  mu[d]  = m;
  scl[d] = gamma[d] * rsqrtf(var + 1e-5f);
}

// ---- voxel scatter: fixed-point int32 atomics => deterministic -------------
__global__ void scatter_kernel(const float* __restrict__ feats, const int* __restrict__ vox,
                               int* __restrict__ acc32, int* __restrict__ cnt) {
  const int r = blockIdx.x;
  const int d = threadIdx.x;
  const int b  = r / P_;
  const int v0 = vox[(size_t)r * 3 + 0];
  const int v1 = vox[(size_t)r * 3 + 1];
  const int v2 = vox[(size_t)r * 3 + 2];
  const int lin = ((b * GX_ + v0) * GY_ + v1) * GZ_ + v2;
  const float f = feats[(size_t)r * D_ + d];
  atomicAdd(&acc32[(size_t)lin * D_ + d], __float2int_rn(f * FIXS));
  if (d == 0) atomicAdd(&cnt[lin], 1);
}

// ---- voxel mean + BN affine + coordinates ----------------------------------
__global__ void finalize_kernel(const int* __restrict__ acc32, const int* __restrict__ cnt,
                                const float* __restrict__ mu, const float* __restrict__ scl,
                                const float* __restrict__ beta,
                                float* __restrict__ out_feats, int* __restrict__ out_coors) {
  const int v = blockIdx.x;
  const int d = threadIdx.x;
  const int c = cnt[v];
  float o = 0.f;
  if (c > 0) {
    const float s = (float)acc32[(size_t)v * D_ + d] * INV_FIXS;
    o = (s / (float)c - mu[d]) * scl[d] + beta[d];
  }
  out_feats[(size_t)v * D_ + d] = o;
  if (d < 4) {
    const int per = GX_ * GY_ * GZ_;
    const int bb  = v / per;
    const int rr  = v - bb * per;
    int val;
    if      (d == 0) val = bb;
    else if (d == 1) val = rr / (GY_ * GZ_);
    else if (d == 2) val = (rr % (GY_ * GZ_)) / GZ_;
    else             val = rr % GZ_;
    out_coors[(size_t)v * 4 + d] = val;
  }
}

// ---------------------------------------------------------------------------
extern "C" void kernel_launch(void* const* d_in, const int* in_sizes, int n_in,
                              void* d_out, int out_size, void* d_ws, size_t ws_size,
                              hipStream_t stream) {
  (void)in_sizes; (void)n_in; (void)out_size; (void)ws_size;
  const float* clip  = (const float*)d_in[0];
  const float* pc    = (const float*)d_in[1];
  const int*   vox   = (const int*)  d_in[2];
  const float* w1    = (const float*)d_in[3];
  const float* w2    = (const float*)d_in[4];
  const float* w3    = (const float*)d_in[5];
  const float* gamma = (const float*)d_in[6];
  const float* beta  = (const float*)d_in[7];

  // ---- workspace layout (byte offsets, 256B aligned chunks) ----
  char* base = (char*)d_ws;
  size_t off = 0;
  auto alloc = [&](size_t bytes) {
    void* p = base + off;
    off += (bytes + 255) & ~(size_t)255;
    return p;
  };
  float* x1    = (float*)alloc((size_t)NIMG * 64 * 64 * 4);      // conv1 out [64][2*32]
  float* x2    = (float*)alloc((size_t)NIMG * 64 * 64 * 4);      // conv2 out [64][64]
  float* x3    = (float*)alloc((size_t)NIMG * D_ * 128 * 4);     // conv3 out [256][128]
  float* F     = (float*)alloc((size_t)NIMG * D_ * 256 * 4);     // fmap row 0
  float* feats = (float*)alloc((size_t)NROWS * D_ * 4);
  float* pS    = (float*)alloc((size_t)NPART * D_ * 4);
  float* pQ    = (float*)alloc((size_t)NPART * D_ * 4);
  float* mu    = (float*)alloc(D_ * 4);
  float* scl   = (float*)alloc(D_ * 4);
  int*   acc32 = (int*)alloc((size_t)NVOX * D_ * 4);
  int*   cnt   = (int*)alloc((size_t)NVOX * 4);
  __bf16* A1  = (__bf16*)alloc((size_t)64 * K1_ * 2);
  __bf16* A2  = (__bf16*)alloc((size_t)64 * K2_ * 2);
  __bf16* A3  = (__bf16*)alloc((size_t)D_ * K3_ * 2);
  __bf16* Bt1 = (__bf16*)alloc((size_t)NIMG * 64 * K1_ * 2);
  __bf16* Bt2 = (__bf16*)alloc((size_t)NIMG * 64 * K2_ * 2);
  __bf16* Bt3 = (__bf16*)alloc((size_t)NIMG * 128 * K3_ * 2);

  // zero voxel accumulators (acc32 and cnt are contiguous)
  hipMemsetAsync(acc32, 0, ((size_t)NVOX * D_ + NVOX) * sizeof(int), stream);

  // weights -> bf16 (K-contiguous in [O][I][ky][kx] order)
  cvt_bf16_kernel<<<(64 * K1_ + 255) / 256, 256, 0, stream>>>(w1, A1, 64 * K1_);
  cvt_bf16_kernel<<<(64 * K2_ + 255) / 256, 256, 0, stream>>>(w2, A2, 64 * K2_);
  cvt_bf16_kernel<<<(D_ * K3_ + 255) / 256, 256, 0, stream>>>(w3, A3, D_ * K3_);

  // conv1
  pack_b1_kernel<<<dim3(64, NIMG), 256, 0, stream>>>(clip, Bt1);
  gemm_relu_kernel<<<dim3(4 * 4, NIMG), 32, 0, stream>>>(A1, Bt1, x1, K1_, 4, 64, 64);
  // conv2
  pack_b2_kernel<<<dim3(64, NIMG), 256, 0, stream>>>(x1, Bt2);
  gemm_relu_kernel<<<dim3(4 * 4, NIMG), 32, 0, stream>>>(A2, Bt2, x2, K2_, 4, 64, 64);
  // conv3
  pack_b3_kernel<<<dim3(128, NIMG), 256, 0, stream>>>(x2, Bt3);
  gemm_relu_kernel<<<dim3(16 * 8, NIMG), 32, 0, stream>>>(A3, Bt3, x3, K3_, 8, 128, D_);

  upsample_row_kernel<<<NIMG * D_, 256, 0, stream>>>(x3, F);
  feats_kernel<<<NROWS, 256, 0, stream>>>(pc, F, feats);
  stats_partial<<<NPART, 256, 0, stream>>>(feats, pS, pQ);
  stats_final<<<1, 256, 0, stream>>>(pS, pQ, gamma, mu, scl);
  scatter_kernel<<<NROWS, 256, 0, stream>>>(feats, vox, acc32, cnt);
  finalize_kernel<<<NVOX, 256, 0, stream>>>(acc32, cnt, mu, scl, beta,
      (float*)d_out, (int*)((float*)d_out + (size_t)NVOX * D_));
}
